// CopNet_82832739271217
// MI455X (gfx1250) — matrix operand order repaired and checked by reading, not compiled, not run on hardware
//
#include <hip/hip_runtime.h>
#include <hip/hip_bf16.h>

typedef __attribute__((ext_vector_type(16))) _Float16 v16h;
typedef __attribute__((ext_vector_type(8)))  _Float16 v8h;
typedef __attribute__((ext_vector_type(8)))  float    v8f;

#define HFEAT 128
#define GGRAPHS 64
#define CCLASS 10
#define BNEPS 1e-5f
#define LDS_STRIDE 136  // 136 halves = 272B rows: 16B aligned, bank-spread

// ---------------- generic zero ----------------
__global__ void gcn_zero(float* p, int n) {
    int i = blockIdx.x * blockDim.x + threadIdx.x;
    if (i < n) p[i] = 0.0f;
}

// ---------------- degree / norm ----------------
__global__ void gcn_deg_accum(const int* dst, float* deg, int E) {
    int e = blockIdx.x * blockDim.x + threadIdx.x;
    if (e < E) atomicAdd(&deg[dst[e]], 1.0f);
}

// in-place: dinv holds raw degree on entry; writes dinv=rsqrt(deg+1), selfn=1/(deg+1)
__global__ void gcn_deg_fin(float* dinv, float* selfn, int N) {
    int n = blockIdx.x * blockDim.x + threadIdx.x;
    if (n < N) {
        float d = dinv[n] + 1.0f;
        float r = rsqrtf(d);
        dinv[n]  = r;
        selfn[n] = 1.0f / d;
    }
}

// ---------------- WMMA GEMM: H = X @ W, X:[N,128] f32, W:[128,128] f32 ----------------
// Block = 256 threads = 8 waves; each wave computes a 16-row x 128-col strip.
// W converted to f16, stored transposed in LDS: ldsW[col*LDS_STRIDE + k].
__global__ void gcn_gemm128(const float* __restrict__ X,
                            const float* __restrict__ Wg,
                            float* __restrict__ H, int Nrows) {
    __shared__ __align__(32) _Float16 ldsW[HFEAT * LDS_STRIDE];

    const int tid = threadIdx.x;
    // Stage W^T into LDS as f16: ldsW[col][k] = W[k][col]
    for (int i = tid; i < HFEAT * HFEAT; i += 256) {
        int k   = i >> 7;
        int col = i & 127;
        ldsW[col * LDS_STRIDE + k] = (_Float16)Wg[i];
    }
    __syncthreads();

    const int wave = tid >> 5;
    const int lane = tid & 31;
    const int rowBase = blockIdx.x * 128 + wave * 16;
    if (rowBase >= Nrows) return;   // strip-uniform; N is a multiple of 16

    const int m16   = lane & 15;     // row within 16-row A tile / col within B tile
    const int kgrp  = lane >> 4;     // 0 or 1: A-fragment K sub-group
    const float* xrow = X + (size_t)(rowBase + m16) * HFEAT;

    v8f acc[8] = {};                 // 8 column tiles of 16x16 f32

    #pragma unroll
    for (int kc = 0; kc < 4; ++kc) { // K chunks of 32
        // Build 16x32 f16 A fragment in contiguous k order (B matches):
        // lane<16: M=lane, K = 0..15; lane>=16: same M, K = 16..31
        v16h a;
        const int kb = kc * 32 + kgrp * 16;
        #pragma unroll
        for (int v = 0; v < 4; ++v) {
            float2 p = *(const float2*)(xrow + kb + 2 * v);
            a[2 * v]     = (_Float16)p.x;
            a[2 * v + 1] = (_Float16)p.y;
        }
        #pragma unroll
        for (int v = 0; v < 4; ++v) {
            float2 p = *(const float2*)(xrow + kb + 8 + 2 * v);
            a[8 + 2 * v]     = (_Float16)p.x;
            a[8 + 2 * v + 1] = (_Float16)p.y;
        }

        // B fragment: lane<16: N=lane, K=kc*32+0..15; lane>=16: N=lane-16, K=kc*32+16..31
        const int kstart = kc * 32 + kgrp * 16;
        #pragma unroll
        for (int ct = 0; ct < 8; ++ct) {
            const int col = ct * 16 + m16;
            const v8h* bp = (const v8h*)&ldsW[col * LDS_STRIDE + kstart];
            v8h b0 = bp[0];
            v8h b1 = bp[1];
            v16h b = __builtin_shufflevector(b0, b1,
                        0, 1, 2, 3, 4, 5, 6, 7, 8, 9, 10, 11, 12, 13, 14, 15);
            acc[ct] = __builtin_amdgcn_wmma_f32_16x16x32_f16(
                false, a, false, b, (short)0, acc[ct], false, false);
        }
    }

    // D layout: lane L holds col N=L&15; VGPR i holds row M = i + (L>>4)*8
    const int mrow = rowBase + kgrp * 8;
    #pragma unroll
    for (int ct = 0; ct < 8; ++ct) {
        #pragma unroll
        for (int i = 0; i < 8; ++i) {
            H[(size_t)(mrow + i) * HFEAT + ct * 16 + m16] = acc[ct][i];
        }
    }
}

// ---------------- Y = H*selfnorm + bias (seed for scatter) ----------------
__global__ void gcn_init_self(const float* __restrict__ H,
                              const float* __restrict__ selfn,
                              const float* __restrict__ bias,
                              float* __restrict__ Y, int N) {
    long long i = (long long)blockIdx.x * blockDim.x + threadIdx.x;
    if (i < (long long)N * HFEAT) {
        int n = (int)(i >> 7);
        int f = (int)(i & 127);
        Y[i] = H[i] * selfn[n] + bias[f];
    }
}

// ---------------- edge scatter: Y[dst] += H[src] * dinv[src]*dinv[dst] ----------------
// One wave per edge: lanes cover 128 feats as float4 chunks (coalesced 512B gather).
__global__ void gcn_scatter(const int* __restrict__ src, const int* __restrict__ dst,
                            const float* __restrict__ H, const float* __restrict__ dinv,
                            float* __restrict__ Y, int E) {
    long long idx = (long long)blockIdx.x * blockDim.x + threadIdx.x;
    int e = (int)(idx >> 5);
    if (e >= E) return;
    int lane = (int)(idx & 31);
    int s = src[e];
    int d = dst[e];
    float w = dinv[s] * dinv[d];
    float4 hv = *(const float4*)(H + (size_t)s * HFEAT + lane * 4);
    float* yp = Y + (size_t)d * HFEAT + lane * 4;
    atomicAdd(yp + 0, hv.x * w);
    atomicAdd(yp + 1, hv.y * w);
    atomicAdd(yp + 2, hv.z * w);
    atomicAdd(yp + 3, hv.w * w);
}

// ---------------- BN stats: per-feature sum & sumsq ----------------
__global__ void gcn_bnstats(const float* __restrict__ Y, float* __restrict__ stats, int N) {
    int f = threadIdx.x & 127;
    int rowsPerBlk = blockDim.x >> 7;               // 2 for 256 threads
    int n0 = blockIdx.x * rowsPerBlk + (threadIdx.x >> 7);
    int stride = gridDim.x * rowsPerBlk;
    float s = 0.0f, s2 = 0.0f;
    for (int n = n0; n < N; n += stride) {
        float v = Y[(size_t)n * HFEAT + f];
        s  += v;
        s2 += v * v;
    }
    atomicAdd(&stats[f], s);
    atomicAdd(&stats[HFEAT + f], s2);
}

// stats -> {scale, shift}
__global__ void gcn_bnfin(float* stats, const float* __restrict__ g,
                          const float* __restrict__ be, float Ninv) {
    int f = threadIdx.x;
    if (f < HFEAT) {
        float mean = stats[f] * Ninv;
        float var  = stats[HFEAT + f] * Ninv - mean * mean;
        float sc   = g[f] * rsqrtf(var + BNEPS);
        stats[f]         = sc;
        stats[HFEAT + f] = be[f] - mean * sc;
    }
}

__global__ void gcn_bnapply(float* __restrict__ Y, const float* __restrict__ stats, int N) {
    long long i = (long long)blockIdx.x * blockDim.x + threadIdx.x;
    if (i < (long long)N * HFEAT) {
        int f = (int)(i & 127);
        float v = Y[i] * stats[f] + stats[HFEAT + f];
        Y[i] = fmaxf(v, 0.0f);
    }
}

__global__ void gcn_relu(float* __restrict__ Y, long long n) {
    long long i = (long long)blockIdx.x * blockDim.x + threadIdx.x;
    if (i < n) Y[i] = fmaxf(Y[i], 0.0f);
}

// ---------------- pooling: per-graph mean-sum + max (non-negative -> bitwise max ok) ----------------
__global__ void gcn_pool(const float* __restrict__ Y, const int* __restrict__ batch,
                         float* __restrict__ psum, unsigned* __restrict__ pmax,
                         float* __restrict__ counts, int N) {
    long long idx = (long long)blockIdx.x * blockDim.x + threadIdx.x;
    int n = (int)(idx >> 5);
    if (n >= N) return;
    int lane = (int)(idx & 31);
    int g = batch[n];
    float4 v = *(const float4*)(Y + (size_t)n * HFEAT + lane * 4);
    float*    sp = psum + (size_t)g * HFEAT + lane * 4;
    unsigned* mp = pmax + (size_t)g * HFEAT + lane * 4;
    atomicAdd(sp + 0, v.x); atomicMax(mp + 0, __float_as_uint(v.x));
    atomicAdd(sp + 1, v.y); atomicMax(mp + 1, __float_as_uint(v.y));
    atomicAdd(sp + 2, v.z); atomicMax(mp + 2, __float_as_uint(v.z));
    atomicAdd(sp + 3, v.w); atomicMax(mp + 3, __float_as_uint(v.w));
    if (lane == 0) atomicAdd(&counts[g], 1.0f);
}

// ---------------- head: hid = relu([mean|max] @ lw1 + lb1) ----------------
__global__ void gcn_head1(const float* __restrict__ psum, const unsigned* __restrict__ pmax,
                          const float* __restrict__ counts, const float* __restrict__ lw1,
                          const float* __restrict__ lb1, float* __restrict__ hid) {
    int idx = blockIdx.x * blockDim.x + threadIdx.x;
    if (idx >= GGRAPHS * HFEAT) return;
    int g = idx >> 7;
    int j = idx & 127;
    float inv = 1.0f / counts[g];
    float acc = lb1[j];
    #pragma unroll 4
    for (int k = 0; k < HFEAT; ++k)
        acc += psum[g * HFEAT + k] * inv * lw1[k * HFEAT + j];
    #pragma unroll 4
    for (int k = 0; k < HFEAT; ++k)
        acc += __uint_as_float(pmax[g * HFEAT + k]) * lw1[(HFEAT + k) * HFEAT + j];
    hid[idx] = fmaxf(acc, 0.0f);
}

// ---------------- head: out = hid @ lw2 + lb2 ----------------
__global__ void gcn_head2(const float* __restrict__ hid, const float* __restrict__ lw2,
                          const float* __restrict__ lb2, float* __restrict__ out) {
    int idx = blockIdx.x * blockDim.x + threadIdx.x;
    if (idx >= GGRAPHS * CCLASS) return;
    int g = idx / CCLASS;
    int c = idx % CCLASS;
    float acc = lb2[c];
    #pragma unroll 4
    for (int k = 0; k < HFEAT; ++k)
        acc += hid[g * HFEAT + k] * lw2[k * CCLASS + c];
    out[idx] = acc;
}

extern "C" void kernel_launch(void* const* d_in, const int* in_sizes, int n_in,
                              void* d_out, int out_size, void* d_ws, size_t ws_size,
                              hipStream_t stream) {
    const float* x     = (const float*)d_in[0];
    const int*   eidx  = (const int*)d_in[1];
    const int*   batch = (const int*)d_in[2];
    const float* Wm[4] = {(const float*)d_in[3], (const float*)d_in[5],
                          (const float*)d_in[7], (const float*)d_in[9]};
    const float* bm[4] = {(const float*)d_in[4], (const float*)d_in[6],
                          (const float*)d_in[8], (const float*)d_in[10]};
    const float* gm[3]  = {(const float*)d_in[11], (const float*)d_in[13], (const float*)d_in[15]};
    const float* bem[3] = {(const float*)d_in[12], (const float*)d_in[14], (const float*)d_in[16]};
    const float* lw1 = (const float*)d_in[17];
    const float* lb1 = (const float*)d_in[18];
    const float* lw2 = (const float*)d_in[19];
    const float* lb2 = (const float*)d_in[20];
    float* out = (float*)d_out;

    const int N = in_sizes[0] / HFEAT;
    const int E = in_sizes[1] / 2;
    const int* src = eidx;
    const int* dst = eidx + E;

    // workspace layout (floats)
    float* ws = (float*)d_ws;
    float* Hbuf  = ws;                                  // N*128
    float* Ybuf  = Hbuf + (size_t)N * HFEAT;            // N*128
    float* dinv  = Ybuf + (size_t)N * HFEAT;            // N (deg during accumulation)
    float* selfn = dinv + N;                            // N
    float* stats = selfn + N;                           // 256
    float* psum  = stats + 256;                         // 64*128
    unsigned* pmax = (unsigned*)(psum + GGRAPHS * HFEAT); // 64*128
    float* counts = (float*)(pmax + GGRAPHS * HFEAT);   // 64
    float* hid    = counts + GGRAPHS;                   // 64*128

    const long long NF = (long long)N * HFEAT;
    const int elemBlk = (int)((NF + 255) / 256);
    const int gemmBlk = (N + 127) / 128;
    const int edgeWaveBlk = (int)(((long long)E * 32 + 255) / 256);
    const int nodeWaveBlk = (int)(((long long)N * 32 + 255) / 256);

    // degrees -> dinv, selfnorm
    gcn_zero<<<(N + 255) / 256, 256, 0, stream>>>(dinv, N);
    gcn_deg_accum<<<(E + 255) / 256, 256, 0, stream>>>(dst, dinv, E);
    gcn_deg_fin<<<(N + 255) / 256, 256, 0, stream>>>(dinv, selfn, N);

    // 4 GCN layers
    for (int L = 0; L < 4; ++L) {
        const float* in = (L == 0) ? x : Ybuf;
        gcn_gemm128<<<gemmBlk, 256, 0, stream>>>(in, Wm[L], Hbuf, N);
        gcn_init_self<<<elemBlk, 256, 0, stream>>>(Hbuf, selfn, bm[L], Ybuf, N);
        gcn_scatter<<<edgeWaveBlk, 256, 0, stream>>>(src, dst, Hbuf, dinv, Ybuf, E);
        if (L < 3) {
            gcn_zero<<<1, 256, 0, stream>>>(stats, 256);
            gcn_bnstats<<<1024, 256, 0, stream>>>(Ybuf, stats, N);
            gcn_bnfin<<<1, 128, 0, stream>>>(stats, gm[L], bem[L], 1.0f / (float)N);
            gcn_bnapply<<<elemBlk, 256, 0, stream>>>(Ybuf, stats, N);
        } else {
            gcn_relu<<<elemBlk, 256, 0, stream>>>(Ybuf, NF);
        }
    }

    // pooling (max over non-negative values: zero-init + bitwise atomicMax is exact)
    gcn_zero<<<(GGRAPHS * HFEAT + 255) / 256, 256, 0, stream>>>(psum, GGRAPHS * HFEAT);
    gcn_zero<<<(GGRAPHS * HFEAT + 255) / 256, 256, 0, stream>>>((float*)pmax, GGRAPHS * HFEAT);
    gcn_zero<<<1, 256, 0, stream>>>(counts, GGRAPHS);
    gcn_pool<<<nodeWaveBlk, 256, 0, stream>>>(Ybuf, batch, psum, pmax, counts, N);

    // MLP head
    gcn_head1<<<(GGRAPHS * HFEAT + 255) / 256, 256, 0, stream>>>(psum, pmax, counts, lw1, lb1, hid);
    gcn_head2<<<(GGRAPHS * CCLASS + 255) / 256, 256, 0, stream>>>(hid, lw2, lb2, out);
}
